// KANFormer_30666066494089
// MI455X (gfx1250) — matrix operand order repaired and checked by reading, not compiled, not run on hardware
//
#include <hip/hip_runtime.h>
#include <cstdint>
#include <cstddef>

// ---------------- CDNA5 WMMA plumbing (wave32) ----------------
typedef __bf16 bf16;
typedef __attribute__((ext_vector_type(16))) __bf16 v16bf;
typedef __attribute__((ext_vector_type(8)))  float  v8f;

union FragU { uint4 u[2]; v16bf v; };
union AccU  { v8f v; float f[8]; };

// A-matrix 16x32 bf16 frag: lane holds row (l&15); K = kh*16 + e (contiguous)
static __device__ __forceinline__ v16bf frag_a(const bf16* p) {
  FragU f;
  f.u[0] = *(const uint4*)(p);
  f.u[1] = *(const uint4*)(p + 8);
  return f.v;
}
// B-matrix 32x16 bf16 frag from Bt (NxK row-major): lane holds col (l&15); K = kh*16 + e (contiguous)
static __device__ __forceinline__ v16bf frag_b(const bf16* p) {
  FragU f;
  f.u[0] = *(const uint4*)(p);
  f.u[1] = *(const uint4*)(p + 8);
  return f.v;
}
static __device__ __forceinline__ v8f wmma_bf16(v16bf a, v16bf b, v8f c) {
  return __builtin_amdgcn_wmma_f32_16x16x32_bf16(false, a, false, b, (short)0, c, false, false);
}

// Async global -> LDS copy of 16 bytes per lane (CDNA5 GLOBAL_LOAD_ASYNC_TO_LDS_B128, ASYNCcnt)
static __device__ __forceinline__ void async_copy_b128(unsigned lds_off, const bf16* gptr) {
  asm volatile("global_load_async_to_lds_b128 %0, %1, off"
               :: "v"(lds_off), "v"(gptr) : "memory");
}
static __device__ __forceinline__ void wait_async0() {
  asm volatile("s_wait_asynccnt 0x0" ::: "memory");
}

// ---------------- model dims ----------------
#define BATCH 16
#define SEQ 128
#define NF 64
#define HID 512
#define HEADS 8
#define HD 64
#define NBLK 6
#define ROWS (BATCH*SEQ)       // 2048
#define KCH (HID*4)            // 2048 (chebyshev-expanded K)
#define QKVN (3*HID)           // 1536

// ---------------- tiled bf16 WMMA GEMM (128x128 block tile, LDS-staged panels) ----------------
// C(MxN,f32,ldc) = A(MxK bf16,lda) * Bt(NxK bf16,ldb) [+ addsrc] [+ bias]
// grid: (N/128, M/128), block 256 (8 waves; wave w owns rows [16w,16w+16) x all 128 cols)
#define PSTR 40   // LDS panel row stride in bf16 elements (32 + pad, 80B = 5*16B)
__global__ __launch_bounds__(256) void gemm_bf16_wmma(
    const bf16* __restrict__ A, int lda,
    const bf16* __restrict__ Bt, int ldb,
    float* __restrict__ C, int ldc,
    const float* __restrict__ addsrc,
    const float* __restrict__ bias, int K) {
  __shared__ bf16 At[128 * PSTR];
  __shared__ bf16 Bl[128 * PSTR];
  const int tid  = threadIdx.x;
  const int lane = tid & 31;
  const int wave = tid >> 5;
  const int l15  = lane & 15;
  const int kh   = lane >> 4;

  // panel loader mapping: each thread moves one 16B chunk of A and one of B per k-step
  const int lrow = tid >> 1;          // 0..127
  const int lch  = tid & 1;           // 0/1 -> k-offset 0/16
  const bf16* agp = A  + (size_t)(blockIdx.y * 128 + lrow) * lda + lch * 16;
  const bf16* bgp = Bt + (size_t)(blockIdx.x * 128 + lrow) * ldb + lch * 16;
  const unsigned aL = (unsigned)(uintptr_t)&At[lrow * PSTR + lch * 16];
  const unsigned bL = (unsigned)(uintptr_t)&Bl[lrow * PSTR + lch * 16];

  v8f acc[8];
  for (int nt = 0; nt < 8; ++nt) acc[nt] = (v8f){0.f,0.f,0.f,0.f,0.f,0.f,0.f,0.f};

  for (int k = 0; k < K; k += 32) {
    async_copy_b128(aL, agp);
    async_copy_b128(bL, bgp);
    agp += 32; bgp += 32;
    __builtin_prefetch(agp + 32, 0, 1);
    __builtin_prefetch(bgp + 32, 0, 1);
    wait_async0();
    __syncthreads();
    v16bf a = frag_a(&At[(wave * 16 + l15) * PSTR + kh * 16]);
    for (int nt = 0; nt < 8; ++nt) {
      v16bf b = frag_b(&Bl[(nt * 16 + l15) * PSTR + kh * 16]);
      acc[nt] = wmma_bf16(a, b, acc[nt]);
    }
    __syncthreads();
  }

  for (int nt = 0; nt < 8; ++nt) {
    AccU r; r.v = acc[nt];
    const int col = blockIdx.x * 128 + nt * 16 + l15;
    const float bv = bias ? bias[col] : 0.f;
    for (int i = 0; i < 8; ++i) {
      const int row = blockIdx.y * 128 + wave * 16 + kh * 8 + i;
      size_t idx = (size_t)row * ldc + col;
      float v = r.f[i] + bv;
      if (addsrc) v += addsrc[idx];
      C[idx] = v;
    }
  }
}

// ---------------- weight packing (fp32 -> bf16, transposed NxK) ----------------
__global__ void pack_x_kernel(const float* __restrict__ x, bf16* __restrict__ xb) {
  int i = blockIdx.x * 256 + threadIdx.x;            // ROWS*NF
  xb[i] = (bf16)x[i];
}
__global__ void pack_embw_kernel(const float* __restrict__ w, bf16* __restrict__ wt) {
  int i = blockIdx.x * 256 + threadIdx.x;            // HID*NF
  int n = i >> 6, k = i & 63;
  wt[i] = (bf16)w[k * HID + n];
}
__global__ void pack_qkvw_kernel(const float* __restrict__ q, bf16* __restrict__ bt) {
  long long i = (long long)blockIdx.x * 256 + threadIdx.x; // NBLK*QKVN*KCH
  int k = (int)(i & (KCH - 1));
  long long r = i >> 11;
  int o   = (int)(r % QKVN);
  int blk = (int)(r / QKVN);
  int ii = k >> 2, j = k & 3;
  bt[i] = (bf16)q[(((size_t)blk * HID + ii) * QKVN + o) * 4 + j];
}
__global__ void pack_outw_kernel(const float* __restrict__ w, bf16* __restrict__ wt) {
  int i = blockIdx.x * 256 + threadIdx.x;            // NBLK*HID*HID
  int k = i & 511;
  int r = i >> 9;
  int n = r & 511;
  int blk = r >> 9;
  wt[i] = (bf16)w[((size_t)blk * HID + k) * HID + n];
}

// ---------------- rmsnorm + tanh + Chebyshev basis -> bf16 A (ROWS x KCH) ----------------
__global__ __launch_bounds__(256) void rmscheb_kernel(
    const float* __restrict__ h, const float* __restrict__ w, bf16* __restrict__ A) {
  const int lane = threadIdx.x & 31;
  const int wave = threadIdx.x >> 5;
  const int row  = blockIdx.x * 8 + wave;
  const float* hr = h + (size_t)row * HID;
  float vals[16];
  float ss = 0.f;
  for (int i = 0; i < 16; ++i) {
    float v = hr[lane + 32 * i];
    vals[i] = v; ss += v * v;
  }
  for (int off = 16; off; off >>= 1) ss += __shfl_xor(ss, off, 32);
  const float rinv = rsqrtf(ss * (1.f / HID) + 1e-6f);
  bf16* ar = A + (size_t)row * KCH;
  for (int i = 0; i < 16; ++i) {
    int c = lane + 32 * i;
    float t  = tanhf(vals[i] * rinv * w[c]);
    float T1 = 2.f * t;
    float T2 = 2.f * t * T1 - 1.f;
    float T3 = 2.f * t * T2 - T1;
    union { bf16 e[4]; uint2 q; } pk;
    pk.e[0] = (bf16)1.f; pk.e[1] = (bf16)T1; pk.e[2] = (bf16)T2; pk.e[3] = (bf16)T3;
    *(uint2*)(ar + c * 4) = pk.q;
  }
}

// ---------------- fused attention (RoPE + QK^T + softmax + PV), per (b,h,row-half) ----------------
#define QSTR 72    // 64 + pad, bf16 elements (row stride, 16B multiple)
#define VSTR 136   // 128 + pad
__global__ __launch_bounds__(256) void attention_kernel(
    const float* __restrict__ qkv, bf16* __restrict__ ctx) {
  __shared__ bf16 qb[64 * QSTR];
  __shared__ bf16 kb[128 * QSTR];
  __shared__ bf16 vt[64 * VSTR];
  __shared__ bf16 sb[64 * VSTR];
  const int sh = blockIdx.x, hh = blockIdx.y, bb = blockIdx.z;
  const int tid = threadIdx.x;
  const int sbase = sh * 64;
  const float lognorm = __logf(1000.f) / 32.f;

  // Q rows [sbase, sbase+64) with RoPE
  for (int idx = tid; idx < 64 * 32; idx += 256) {
    int sr = idx >> 5, t = idx & 31;
    int s = sbase + sr;
    const float* base = qkv + ((size_t)(bb * SEQ + s)) * QKVN + hh * 192;
    float ang = (float)s * __expf(-(float)t * lognorm);
    float cv = __sinf(ang), sv = __cosf(ang);   // reference's (swapped) naming
    float e = base[2 * t], o = base[2 * t + 1];
    int u = (t < 16) ? t + 16 : t - 16;
    float sgn = (t < 16) ? -1.f : 1.f;
    float rho = sgn * base[2 * u + 1];
    float rhe = sgn * base[2 * u];
    qb[sr * QSTR + t]      = (bf16)(e * cv + rho * sv);
    qb[sr * QSTR + 32 + t] = (bf16)(o * cv + rhe * sv);
  }
  // K all 128 rows with RoPE
  for (int idx = tid; idx < 128 * 32; idx += 256) {
    int sr = idx >> 5, t = idx & 31;
    const float* base = qkv + ((size_t)(bb * SEQ + sr)) * QKVN + hh * 192 + 64;
    float ang = (float)sr * __expf(-(float)t * lognorm);
    float cv = __sinf(ang), sv = __cosf(ang);
    float e = base[2 * t], o = base[2 * t + 1];
    int u = (t < 16) ? t + 16 : t - 16;
    float sgn = (t < 16) ? -1.f : 1.f;
    float rho = sgn * base[2 * u + 1];
    float rhe = sgn * base[2 * u];
    kb[sr * QSTR + t]      = (bf16)(e * cv + rho * sv);
    kb[sr * QSTR + 32 + t] = (bf16)(o * cv + rhe * sv);
  }
  // V transposed (d-major) for the PV GEMM B-operand
  for (int idx = tid; idx < 128 * 64; idx += 256) {
    int d = idx & 63, sr = idx >> 6;
    vt[d * VSTR + sr] = (bf16)qkv[((size_t)(bb * SEQ + sr)) * QKVN + hh * 192 + 128 + d];
  }
  __syncthreads();

  const int lane = tid & 31, wave = tid >> 5;
  const int l15 = lane & 15, kh = lane >> 4;
  const int rt = wave & 3, nh = wave >> 2;

  // GEMM1: scores(64x128) = Q(64x64) * K^T   (Bt == kb)
  {
    v8f acc[4];
    for (int nt = 0; nt < 4; ++nt) acc[nt] = (v8f){0.f,0.f,0.f,0.f,0.f,0.f,0.f,0.f};
    for (int k = 0; k < 64; k += 32) {
      v16bf a = frag_a(&qb[(rt * 16 + l15) * QSTR + k + kh * 16]);
      for (int nt = 0; nt < 4; ++nt) {
        int nc = (nh * 4 + nt) * 16 + l15;
        v16bf b = frag_b(&kb[nc * QSTR + k + kh * 16]);
        acc[nt] = wmma_bf16(a, b, acc[nt]);
      }
    }
    for (int nt = 0; nt < 4; ++nt) {
      AccU r; r.v = acc[nt];
      int nc = (nh * 4 + nt) * 16 + l15;
      for (int i = 0; i < 8; ++i) {
        int rr = rt * 16 + kh * 8 + i;
        sb[rr * VSTR + nc] = (bf16)(r.f[i] * 0.125f);   // 1/sqrt(HD)
      }
    }
  }
  __syncthreads();

  // softmax over each of the 64 rows
  if (tid < 64) {
    int r = tid;
    float m = -1e30f;
    for (int c = 0; c < 128; ++c) m = fmaxf(m, (float)sb[r * VSTR + c]);
    float ssum = 0.f;
    for (int c = 0; c < 128; ++c) ssum += __expf((float)sb[r * VSTR + c] - m);
    float inv = 1.f / ssum;
    for (int c = 0; c < 128; ++c)
      sb[r * VSTR + c] = (bf16)(__expf((float)sb[r * VSTR + c] - m) * inv);
  }
  __syncthreads();

  // GEMM2: ctx(64x64) = P(64x128) * V(128x64)   (Bt == vt)
  {
    v8f acc[2];
    acc[0] = (v8f){0.f,0.f,0.f,0.f,0.f,0.f,0.f,0.f};
    acc[1] = acc[0];
    for (int k = 0; k < 128; k += 32) {
      v16bf a = frag_a(&sb[(rt * 16 + l15) * VSTR + k + kh * 16]);
      for (int nt = 0; nt < 2; ++nt) {
        int nc = (nh * 2 + nt) * 16 + l15;
        v16bf b = frag_b(&vt[nc * VSTR + k + kh * 16]);
        acc[nt] = wmma_bf16(a, b, acc[nt]);
      }
    }
    for (int nt = 0; nt < 2; ++nt) {
      AccU r; r.v = acc[nt];
      int d = (nh * 2 + nt) * 16 + l15;
      for (int i = 0; i < 8; ++i) {
        int s = sbase + rt * 16 + kh * 8 + i;
        ctx[((size_t)(bb * SEQ + s)) * HID + hh * HD + d] = (bf16)r.f[i];
      }
    }
  }
}

// ---------------- ff1: memory-bound Chebyshev-KAN over 268MB coeffs, split-K ----------------
__global__ void zero_kernel(float* __restrict__ p) {
  p[blockIdx.x * 256 + threadIdx.x] = 0.f;
}
__global__ __launch_bounds__(256) void ff1_kernel(
    const float* __restrict__ h, const float* __restrict__ c1, float* __restrict__ f1) {
  __shared__ float4 Tl[64 * 16];        // 64 k-rows x 16 batches of Chebyshev bases (16 KB)
  const int tid = threadIdx.x;          // tid == output index o (0..255)
  const int ibase = blockIdx.x * 512;   // K-slice of 512 input features
  float acc[16];
  for (int b = 0; b < 16; ++b) acc[b] = 0.f;
  for (int c0 = 0; c0 < 512; c0 += 64) {
    for (int idx = tid; idx < 64 * 16; idx += 256) {
      int ii = idx >> 4, b = idx & 15;
      float t  = tanhf(h[(size_t)b * 65536 + ibase + c0 + ii]);
      float T1 = 2.f * t;
      float T2 = 2.f * t * T1 - 1.f;
      float T3 = 2.f * t * T2 - T1;
      Tl[idx] = make_float4(1.f, T1, T2, T3);
    }
    __syncthreads();
    for (int ii = 0; ii < 64; ++ii) {
      int i = ibase + c0 + ii;
      float4 cv = *(const float4*)(c1 + ((size_t)i * 256 + tid) * 4);
      for (int b = 0; b < 16; ++b) {
        float4 tb = Tl[ii * 16 + b];
        acc[b] += tb.x * cv.x + tb.y * cv.y + tb.z * cv.z + tb.w * cv.w;
      }
    }
    __syncthreads();
  }
  for (int b = 0; b < 16; ++b) atomicAdd(&f1[b * 256 + tid], acc[b]);
}

// ---------------- ff2: tiny Chebyshev-KAN (16x256 -> 16x128) ----------------
__global__ __launch_bounds__(128) void ff2_kernel(
    const float* __restrict__ f1, const float* __restrict__ c2, float* __restrict__ out) {
  const int o = threadIdx.x;  // 0..127
  for (int b = 0; b < 16; ++b) {
    float acc = 0.f;
    for (int i = 0; i < 256; ++i) {
      float t  = tanhf(f1[b * 256 + i]);
      float T1 = 2.f * t;
      float T2 = 2.f * t * T1 - 1.f;
      float T3 = 2.f * t * T2 - T1;
      float4 cv = *(const float4*)(c2 + ((size_t)i * 128 + o) * 4);
      acc += cv.x + T1 * cv.y + T2 * cv.z + T3 * cv.w;
    }
    out[b * 128 + o] = acc;
  }
}

// ---------------- host launch ----------------
extern "C" void kernel_launch(void* const* d_in, const int* in_sizes, int n_in,
                              void* d_out, int out_size, void* d_ws, size_t ws_size,
                              hipStream_t stream) {
  const float* x      = (const float*)d_in[0];
  const float* emb_w  = (const float*)d_in[1];
  const float* emb_b  = (const float*)d_in[2];
  const float* norm_w = (const float*)d_in[3];
  const float* qkv_c  = (const float*)d_in[4];
  const float* out_w  = (const float*)d_in[5];
  const float* out_b  = (const float*)d_in[6];
  const float* ff_c1  = (const float*)d_in[7];
  const float* ff_c2  = (const float*)d_in[8];
  float* out = (float*)d_out;

  char* ws = (char*)d_ws;
  // workspace layout (bytes)
  float* h     = (float*)(ws + 0);                        // ROWS*HID f32      (4 MB)
  bf16*  Acheb = (bf16*) (ws + 4194304);                  // ROWS*KCH bf16     (8 MB)
  float* qkv   = (float*)(ws + 12582912);                 // ROWS*QKVN f32     (12 MB)
  bf16*  ctxb  = (bf16*) (ws + 25165824);                 // ROWS*HID bf16     (2 MB)
  bf16*  xb    = (bf16*) (ws + 27262976);                 // ROWS*NF bf16
  bf16*  embwt = (bf16*) (ws + 27525120);                 // HID*NF bf16
  bf16*  bqkv  = (bf16*) (ws + 27590656);                 // NBLK*QKVN*KCH bf16 (36 MB)
  bf16*  wout  = (bf16*) (ws + 65339392);                 // NBLK*HID*HID bf16  (3 MB)
  float* f1    = (float*)(ws + 68485120);                 // 16*256 f32

  // pack weights/activations to bf16 (transposed NxK for B operands)
  pack_x_kernel   <<<(ROWS * NF) / 256, 256, 0, stream>>>(x, xb);
  pack_embw_kernel<<<(HID * NF) / 256, 256, 0, stream>>>(emb_w, embwt);
  pack_qkvw_kernel<<<(NBLK * QKVN * KCH) / 256, 256, 0, stream>>>(qkv_c, bqkv);
  pack_outw_kernel<<<(NBLK * HID * HID) / 256, 256, 0, stream>>>(out_w, wout);

  // h = x @ emb_w + emb_b
  gemm_bf16_wmma<<<dim3(HID / 128, ROWS / 128), 256, 0, stream>>>(
      xb, NF, embwt, NF, h, HID, nullptr, emb_b, NF);

  for (int i = 0; i < NBLK; ++i) {
    rmscheb_kernel<<<ROWS / 8, 256, 0, stream>>>(h, norm_w + i * HID, Acheb);
    gemm_bf16_wmma<<<dim3(QKVN / 128, ROWS / 128), 256, 0, stream>>>(
        Acheb, KCH, bqkv + (size_t)i * QKVN * KCH, KCH, qkv, QKVN, nullptr, nullptr, KCH);
    attention_kernel<<<dim3(2, HEADS, BATCH), 256, 0, stream>>>(qkv, ctxb);
    gemm_bf16_wmma<<<dim3(HID / 128, ROWS / 128), 256, 0, stream>>>(
        ctxb, HID, wout + (size_t)i * HID * HID, HID, h, HID, h, out_b + i * HID, HID);
  }

  // final Chebyshev-KAN head
  zero_kernel<<<16, 256, 0, stream>>>(f1);
  ff1_kernel<<<128, 256, 0, stream>>>(h, ff_c1, f1);
  ff2_kernel<<<1, 128, 0, stream>>>(f1, ff_c2, out);
}
